// PhysicsInformedAttention_86011015070053
// MI455X (gfx1250) — matrix operand order repaired and checked by reading, not compiled, not run on hardware
//
#include <hip/hip_runtime.h>
#include <hip/hip_bf16.h>

// ---------------------------------------------------------------------------
// PhysicsInformedAttention for MI455X (gfx1250, wave32, WMMA + async-to-LDS)
// B=4, L=1024, D=1024, H=16, dk=64
// ---------------------------------------------------------------------------

#define BB 4
#define LL 1024
#define DD 1024
#define HH 16
#define DK 64

typedef _Float16 v16h __attribute__((ext_vector_type(16)));
typedef _Float16 h8   __attribute__((ext_vector_type(8)));
typedef float    v8f  __attribute__((ext_vector_type(8)));

// ---- fragment helpers ------------------------------------------------------
// A-matrix (16x32 f16): lanes 0-15 row=lane hold K 0..15 contiguous,
// lanes 16-31 hold K 16..31 (caller passes c0 = 8*half; base is 2*c0).
__device__ __forceinline__ v16h fragA(const _Float16* row, int c0) {
    v16h r;
    *(h8*)&r       = *(const h8*)(row + 2 * c0);
    *((h8*)&r + 1) = *(const h8*)(row + 2 * c0 + 8);
    return r;
}
// B-matrix (32x16 f16): lanes 0-15 col=lane hold K 0..15 contiguous,
// lanes 16-31 hold K 16..31 (caller passes the 16-element base pointer).
__device__ __forceinline__ v16h fragB(const _Float16* p) {
    v16h r;
    *(h8*)&r       = *(const h8*)p;
    *((h8*)&r + 1) = *(const h8*)(p + 8);
    return r;
}
__device__ __forceinline__ v8f wmma16(v16h a, v16h b, v8f c) {
    return __builtin_amdgcn_wmma_f32_16x16x32_f16(false, a, false, b,
                                                  (short)0, c, false, false);
}
__device__ __forceinline__ float redmax16(float v) {
    #pragma unroll
    for (int m = 1; m < 16; m <<= 1) v = fmaxf(v, __shfl_xor(v, m, 32));
    return v;
}
__device__ __forceinline__ float redsum16(float v) {
    #pragma unroll
    for (int m = 1; m < 16; m <<= 1) v += __shfl_xor(v, m, 32);
    return v;
}
// gfx1250 async copy: ASYNCcnt-tracked global->LDS 16B move (per-lane).
// VDST operand = 32-bit LDS byte offset (generic LDS address truncates to it).
__device__ __forceinline__ void async_g2l_b128(const _Float16* gptr, void* lptr) {
    unsigned int ldsoff = (unsigned int)(unsigned long long)lptr;
    asm volatile("global_load_async_to_lds_b128 %0, %1, off"
                 :: "v"(ldsoff), "v"(gptr) : "memory");
}
__device__ __forceinline__ void wait_async0() {
    asm volatile("s_wait_asynccnt 0x0" ::: "memory");
}

// ---- small prep kernels ----------------------------------------------------
__global__ void k_cvt_f16(const float* __restrict__ src, _Float16* __restrict__ dst, int n) {
    int i = blockIdx.x * 256 + threadIdx.x;
    if (i < n) dst[i] = (_Float16)src[i];
}

// proj[b,l,h] = sum_s ss[b,l,s] * ssw[h,s]   (B*L*H = 65536 threads)
__global__ void k_proj(const float* __restrict__ ss, const float* __restrict__ ssw,
                       float* __restrict__ proj) {
    int idx = blockIdx.x * 256 + threadIdx.x;
    int h  = idx & 15;
    int bl = idx >> 4;
    float acc = 0.f;
    #pragma unroll
    for (int s = 0; s < 8; ++s) acc += ss[bl * 8 + s] * ssw[h * 8 + s];
    proj[bl * 16 + h] = acc;
}

// bias[b,i,j] = dist_pen + inter + struct (head-independent part)
__global__ void k_bias(const float* __restrict__ pf, const int* __restrict__ aa,
                       const float* __restrict__ proj, const float* __restrict__ inter_mat,
                       const float* __restrict__ pscale, const float* __restrict__ pdecay,
                       float* __restrict__ bias) {
    size_t idx = (size_t)blockIdx.x * 256 + threadIdx.x;   // < B*L*L = 4M
    int j = (int)(idx & 1023);
    int i = (int)((idx >> 10) & 1023);
    int b = (int)(idx >> 20);

    float dist  = pf[idx];
    dist        = fminf(fmaxf(dist, 0.1f), 50.0f);
    float decay = fminf(fmaxf(pdecay[0], 0.1f), 5.0f);
    float sig   = 1.0f / (1.0f + __expf(-pscale[0]));
    float pen   = -__powf(dist, decay) * sig;

    int ai = aa[b * LL + i]; ai = ai < 0 ? 0 : (ai > 24 ? 24 : ai);
    int aj = aa[b * LL + j]; aj = aj < 0 ? 0 : (aj > 24 ? 24 : aj);
    float sym   = 0.5f * (inter_mat[ai * 25 + aj] + inter_mat[aj * 25 + ai]);
    float inter = sym / (1.0f + fabsf((float)(i - j)));

    const float* pi = proj + ((size_t)b * LL + i) * 16;
    const float* pj = proj + ((size_t)b * LL + j) * 16;
    float d = 0.f;
    #pragma unroll
    for (int t = 0; t < 16; ++t) d += pi[t] * pj[t];
    float st = 1.0f / (1.0f + __expf(-d)) - 0.5f;

    bias[idx] = pen + inter + st;
}

// ---- QKV projection GEMM (WMMA + EXEC-uniform async A staging) -------------
// Q = X @ Wq^T etc.  Block = 8 waves computes a 32x512 tile.  A 32x64 k-chunk
// (4 KB) is double-buffered in LDS: all 256 lanes issue exactly one
// global_load_async_to_lds_b128 per staging step (no EXEC divergence), then
// s_wait_asynccnt 0 + barrier; staging of chunk k+1 overlaps 16 WMMAs on k.
// Q,K written head-split (B,H,L,dk); V written transposed (B,H,dk,L).
__global__ __launch_bounds__(256) void k_gemm_qkv(
    const _Float16* __restrict__ xh,
    const _Float16* __restrict__ wqh, const _Float16* __restrict__ wkh,
    const _Float16* __restrict__ wvh,
    _Float16* __restrict__ qh, _Float16* __restrict__ kh, _Float16* __restrict__ vT) {
    __shared__ __align__(16) _Float16 At[2][32][72];   // 64 data + 8 pad halfs/row

    const int tid  = threadIdx.x;
    const int lane = tid & 31;
    const int wv   = tid >> 5;
    const int mat  = blockIdx.x >> 8;          // 0=Q 1=K 2=V  (256 blocks each)
    const int rem  = blockIdx.x & 255;
    const int tm   = rem >> 1;                 // 0..127  (32-row tiles)
    const int tn   = rem & 1;                  // 0..1    (512-col tiles)
    const int hs   = lane >> 4, lr = lane & 15;
    const int n0   = tn * 512 + wv * 64;

    const _Float16* W = (mat == 0) ? wqh : (mat == 1) ? wkh : wvh;

    // staging: 32 rows x 128B; all 256 tids move one 16B segment per step
    const int srow = tid >> 3, sseg = (tid & 7) * 8;
    const _Float16* sgp = xh + (size_t)(tm * 32 + srow) * DD + sseg;

    async_g2l_b128(sgp, &At[0][srow][sseg]);
    wait_async0();
    __syncthreads();

    v8f acc[2][4] = {};
    for (int kk0 = 0; kk0 < DD; kk0 += 64) {
        const int cur = (kk0 >> 6) & 1;
        if (kk0 + 64 < DD)                      // uniform branch, no EXEC games
            async_g2l_b128(sgp + kk0 + 64, &At[cur ^ 1][srow][sseg]);

        #pragma unroll
        for (int sub = 0; sub < 64; sub += 32) {
            v16h a0 = fragA(&At[cur][lr][sub],      hs * 8);
            v16h a1 = fragA(&At[cur][16 + lr][sub], hs * 8);
            #pragma unroll
            for (int ns = 0; ns < 4; ++ns) {
                v16h b = fragB(W + (size_t)(n0 + ns * 16 + lr) * DD + kk0 + sub + hs * 16);
                acc[0][ns] = wmma16(a0, b, acc[0][ns]);
                acc[1][ns] = wmma16(a1, b, acc[1][ns]);
            }
        }
        if (kk0 + 64 < DD) {
            wait_async0();
            __syncthreads();
        }
    }

    #pragma unroll
    for (int mb = 0; mb < 2; ++mb)
        #pragma unroll
        for (int ns = 0; ns < 4; ++ns)
            #pragma unroll
            for (int r = 0; r < 8; ++r) {
                int gm = tm * 32 + mb * 16 + r + hs * 8;
                int gn = n0 + ns * 16 + lr;
                int b  = gm >> 10, l = gm & 1023;
                int h  = gn >> 6,  d = gn & 63;
                _Float16 v = (_Float16)acc[mb][ns][r];
                if (mat == 0)      qh[(((size_t)(b * HH + h)) * LL + l) * DK + d] = v;
                else if (mat == 1) kh[(((size_t)(b * HH + h)) * LL + l) * DK + d] = v;
                else               vT[(((size_t)(b * HH + h)) * DK + d) * LL + l] = v;
            }
}

// ---- fused flash attention (WMMA) -----------------------------------------
// one wave per (b, h, 16-row query tile); online softmax over L keys in 32-chunks.
__global__ __launch_bounds__(256) void k_attn(
    const _Float16* __restrict__ qh, const _Float16* __restrict__ kh,
    const _Float16* __restrict__ vT, const float* __restrict__ bias,
    const float* __restrict__ dist_bias, _Float16* __restrict__ ctxh) {
    __shared__ __align__(16) _Float16 Pst[8][16][40]; // per-wave P staging (C->A layout)
    const int lane = threadIdx.x & 31;
    const int wvid = threadIdx.x >> 5;
    const int w    = blockIdx.x * 8 + wvid;    // < 4096
    const int b    = w >> 10;
    const int rem  = w & 1023;
    const int h    = rem >> 6;
    const int q0   = (rem & 63) << 4;
    const int hs   = lane >> 4, lr = lane & 15;

    const _Float16* qrow  = qh + (((size_t)(b * HH + h)) * LL + q0 + lr) * DK;
    const _Float16* kbase = kh + ((size_t)(b * HH + h)) * LL * DK;
    const _Float16* vbase = vT + ((size_t)(b * HH + h)) * DK * LL;
    const float*    biasb = bias + (size_t)b * LL * LL;
    const float     db    = dist_bias[h];

    v16h aQ0 = fragA(qrow, 0  + hs * 8);
    v16h aQ1 = fragA(qrow, 32 + hs * 8);

    float mrun[8], lrun[8];
    #pragma unroll
    for (int r = 0; r < 8; ++r) { mrun[r] = -1e30f; lrun[r] = 0.f; }
    v8f acc0 = {}, acc1 = {}, acc2 = {}, acc3 = {};

    for (int kb = 0; kb < LL; kb += 32) {
        const _Float16* krow0 = kbase + (size_t)(kb + lr) * DK;
        const _Float16* krow1 = kbase + (size_t)(kb + 16 + lr) * DK;
        if (kb + 32 < LL) {                     // gfx1250 global_prefetch path
            __builtin_prefetch(kbase + (size_t)(kb + 32 + lr) * DK, 0, 1);
            __builtin_prefetch(vbase + (size_t)lr * LL + kb + 32, 0, 1);
        }
        v8f s0 = {}, s1 = {};
        s0 = wmma16(aQ0, fragB(krow0 + hs * 16), s0);
        s0 = wmma16(aQ1, fragB(krow0 + 32 + hs * 16), s0);
        s1 = wmma16(aQ0, fragB(krow1 + hs * 16), s1);
        s1 = wmma16(aQ1, fragB(krow1 + 32 + hs * 16), s1);

        const int i0 = q0 + hs * 8;
        const int j0 = kb + lr;
        #pragma unroll
        for (int r = 0; r < 8; ++r) {
            const float* bp = biasb + (size_t)(i0 + r) * LL + j0;
            s0[r] = s0[r] * 0.125f + bp[0]  + db;   // 1/sqrt(dk)=0.125
            s1[r] = s1[r] * 0.125f + bp[16] + db;
        }
        #pragma unroll
        for (int r = 0; r < 8; ++r) {
            float mx    = redmax16(fmaxf(s0[r], s1[r]));
            float mnew  = fmaxf(mrun[r], mx);
            float alpha = __expf(mrun[r] - mnew);
            float p0    = __expf(s0[r] - mnew);
            float p1    = __expf(s1[r] - mnew);
            float rs    = redsum16(p0 + p1);
            lrun[r] = lrun[r] * alpha + rs;
            mrun[r] = mnew;
            acc0[r] *= alpha; acc1[r] *= alpha; acc2[r] *= alpha; acc3[r] *= alpha;
            Pst[wvid][r + hs * 8][lr]      = (_Float16)p0;
            Pst[wvid][r + hs * 8][lr + 16] = (_Float16)p1;
        }
        // P tile: 16x32 A-fragment (same-wave DS ops are in-order)
        v16h aP = fragA(&Pst[wvid][lr][0], hs * 8);
        const _Float16* vk = vbase + kb + hs * 16;
        acc0 = wmma16(aP, fragB(vk + (size_t)(0 * 16 + lr) * LL), acc0);
        acc1 = wmma16(aP, fragB(vk + (size_t)(1 * 16 + lr) * LL), acc1);
        acc2 = wmma16(aP, fragB(vk + (size_t)(2 * 16 + lr) * LL), acc2);
        acc3 = wmma16(aP, fragB(vk + (size_t)(3 * 16 + lr) * LL), acc3);
    }
    #pragma unroll
    for (int r = 0; r < 8; ++r) {
        float inv = 1.0f / lrun[r];
        size_t gm = (size_t)b * LL + q0 + r + hs * 8;
        _Float16* orow = ctxh + gm * DD + h * DK + lr;
        orow[0]  = (_Float16)(acc0[r] * inv);
        orow[16] = (_Float16)(acc1[r] * inv);
        orow[32] = (_Float16)(acc2[r] * inv);
        orow[48] = (_Float16)(acc3[r] * inv);
    }
}

// ---- output GEMM (WMMA + EXEC-uniform async A staging): ctx @ Wo^T + bo ----
__global__ __launch_bounds__(256) void k_gemm_out(
    const _Float16* __restrict__ ctxh, const _Float16* __restrict__ woh,
    const float* __restrict__ bo, float* __restrict__ out) {
    __shared__ __align__(16) _Float16 At[2][32][72];

    const int tid  = threadIdx.x;
    const int lane = tid & 31;
    const int wv   = tid >> 5;
    const int tm   = blockIdx.x >> 1;          // 0..127
    const int tn   = blockIdx.x & 1;           // 0..1
    const int hs   = lane >> 4, lr = lane & 15;
    const int n0   = tn * 512 + wv * 64;

    const int srow = tid >> 3, sseg = (tid & 7) * 8;
    const _Float16* sgp = ctxh + (size_t)(tm * 32 + srow) * DD + sseg;

    async_g2l_b128(sgp, &At[0][srow][sseg]);
    wait_async0();
    __syncthreads();

    v8f acc[2][4] = {};
    for (int kk0 = 0; kk0 < DD; kk0 += 64) {
        const int cur = (kk0 >> 6) & 1;
        if (kk0 + 64 < DD)
            async_g2l_b128(sgp + kk0 + 64, &At[cur ^ 1][srow][sseg]);

        #pragma unroll
        for (int sub = 0; sub < 64; sub += 32) {
            v16h a0 = fragA(&At[cur][lr][sub],      hs * 8);
            v16h a1 = fragA(&At[cur][16 + lr][sub], hs * 8);
            #pragma unroll
            for (int ns = 0; ns < 4; ++ns) {
                v16h b = fragB(woh + (size_t)(n0 + ns * 16 + lr) * DD + kk0 + sub + hs * 16);
                acc[0][ns] = wmma16(a0, b, acc[0][ns]);
                acc[1][ns] = wmma16(a1, b, acc[1][ns]);
            }
        }
        if (kk0 + 64 < DD) {
            wait_async0();
            __syncthreads();
        }
    }

    #pragma unroll
    for (int ns = 0; ns < 4; ++ns) {
        const int   gn  = n0 + ns * 16 + lr;
        const float bov = bo[gn];
        #pragma unroll
        for (int mb = 0; mb < 2; ++mb)
            #pragma unroll
            for (int r = 0; r < 8; ++r) {
                int gm = tm * 32 + mb * 16 + r + hs * 8;
                out[(size_t)gm * DD + gn] = acc[mb][ns][r] + bov;
            }
    }
}

// ---------------------------------------------------------------------------
extern "C" void kernel_launch(void* const* d_in, const int* in_sizes, int n_in,
                              void* d_out, int out_size, void* d_ws, size_t ws_size,
                              hipStream_t stream) {
    (void)in_sizes; (void)n_in; (void)out_size; (void)ws_size;
    const float* x      = (const float*)d_in[0];
    const float* pf     = (const float*)d_in[1];
    const int*   aa     = (const int*)  d_in[2];
    const float* ss     = (const float*)d_in[3];
    const float* Wq     = (const float*)d_in[4];
    const float* Wk     = (const float*)d_in[5];
    const float* Wv     = (const float*)d_in[6];
    const float* Wo     = (const float*)d_in[7];
    const float* bo     = (const float*)d_in[8];
    const float* dbias  = (const float*)d_in[9];
    const float* imat   = (const float*)d_in[10];
    const float* ssw    = (const float*)d_in[11];
    const float* pscale = (const float*)d_in[12];
    const float* pdecay = (const float*)d_in[13];
    float*       out    = (float*)d_out;

    char* ws = (char*)d_ws;
    const size_t MB = (size_t)1 << 20;
    _Float16* xh   = (_Float16*)(ws + 0);          //  8 MB  (B*L, D) f16
    _Float16* wqh  = (_Float16*)(ws + 8  * MB);    //  2 MB
    _Float16* wkh  = (_Float16*)(ws + 10 * MB);    //  2 MB
    _Float16* wvh  = (_Float16*)(ws + 12 * MB);    //  2 MB
    _Float16* woh  = (_Float16*)(ws + 14 * MB);    //  2 MB
    _Float16* qh   = (_Float16*)(ws + 16 * MB);    //  8 MB  (B,H,L,dk)
    _Float16* kh   = (_Float16*)(ws + 24 * MB);    //  8 MB  (B,H,L,dk)
    _Float16* vT   = (_Float16*)(ws + 32 * MB);    //  8 MB  (B,H,dk,L)
    _Float16* ctxh = (_Float16*)(ws + 40 * MB);    //  8 MB  (B*L, D)
    float*    bias = (float*)   (ws + 48 * MB);    // 16 MB  (B,L,L)
    float*    proj = (float*)   (ws + 64 * MB);    // 256 KB (B,L,H)

    const int nX = BB * LL * DD;   // 4M
    const int nW = DD * DD;        // 1M
    k_cvt_f16<<<(nX + 255) / 256, 256, 0, stream>>>(x,  xh,  nX);
    k_cvt_f16<<<(nW + 255) / 256, 256, 0, stream>>>(Wq, wqh, nW);
    k_cvt_f16<<<(nW + 255) / 256, 256, 0, stream>>>(Wk, wkh, nW);
    k_cvt_f16<<<(nW + 255) / 256, 256, 0, stream>>>(Wv, wvh, nW);
    k_cvt_f16<<<(nW + 255) / 256, 256, 0, stream>>>(Wo, woh, nW);

    k_proj<<<(BB * LL * HH) / 256, 256, 0, stream>>>(ss, ssw, proj);
    k_bias<<<(BB * LL * LL) / 256, 256, 0, stream>>>(pf, aa, proj, imat, pscale, pdecay, bias);

    // 3 matrices * 128 m-tiles * 2 n-tiles
    k_gemm_qkv<<<3 * 256, 256, 0, stream>>>(xh, wqh, wkh, wvh, qh, kh, vT);

    // B*H*(L/16) = 4096 waves / 8 per block
    k_attn<<<4096 / 8, 256, 0, stream>>>(qh, kh, vT, bias, dbias, ctxh);

    k_gemm_out<<<256, 256, 0, stream>>>(ctxh, woh, bo, out);
}
